// WaterNet0119_41583873359967
// MI455X (gfx1250) — matrix operand order repaired and checked by reading, not compiled, not run on hardware
//
#include <hip/hip_runtime.h>
#include <math.h>

// ---------------- problem constants ----------------
#define NT   365
#define NS   256
#define NH   256
#define NG   32
#define NR   16
#define HID  256
#define M_ROWS (NT*NS)        // 93440 (divisible by 16)
#define K1PAD 64              // 38 padded to 64 for two K=32 WMMA chunks
#define N2   768              // NH*3
#define T_OUT (NT-NR+1)       // 350

typedef __attribute__((ext_vector_type(16))) _Float16 v16h;
typedef __attribute__((ext_vector_type(8)))  _Float16 v8h;
typedef __attribute__((ext_vector_type(8)))  float    v8f;

// ---------------- helpers ----------------
__device__ inline float sigm(float x) { return 1.0f / (1.0f + expf(-x)); }

// A-fragment (16-bit A 16x32 layout): lane m=lane&15, kb=(lane>>4)*8.
// halfs 0..7 -> K=kb..kb+7 (contiguous), halfs 8..15 -> K=16+kb..16+kb+7.
__device__ inline v16h ld_a_frag(const _Float16* p) {
  v8h lo = *(const v8h*)p;
  v8h hi = *(const v8h*)(p + 16);
  v16h r;
#pragma unroll
  for (int i = 0; i < 8; ++i) { r[i] = lo[i]; r[i + 8] = hi[i]; }
  return r;
}

// B-fragment (B is KxN, per lane n=lane&15, kb=(lane>>4)*8): halfs 0..7 -> K=kb..kb+7, halfs 8..15 -> K=16+kb..16+kb+7.
__device__ inline v16h ld_b_frag(const _Float16* p) {
  v8h lo = *(const v8h*)p;
  v8h hi = *(const v8h*)(p + 16);
  v16h r;
#pragma unroll
  for (int i = 0; i < 8; ++i) { r[i] = lo[i]; r[i + 8] = hi[i]; }
  return r;
}

__device__ inline float snow_frac(float Ta, float Tb) {
  float ratio = (Ta + Tb) / (Tb - Ta);
  ratio = fminf(fmaxf(ratio, -1.0f), 1.0f);
  float vf = acosf(ratio) * (1.0f / 3.1415f);
  if (Ta >= 0.0f) vf = 0.0f;
  if (Tb <= 0.0f) vf = 1.0f;
  return vf;
}

// ---------------- prep kernels (f32 -> f16 packing/transpose) ----------------
// xcT padded to K1PAD halfs per row: [x(6) | xc(32) | zeros]
__global__ void k_prep_xcT(const float* __restrict__ x, const float* __restrict__ xc,
                           _Float16* __restrict__ xp) {
  int idx = blockIdx.x * 256 + threadIdx.x;          // M_ROWS*K1PAD
  int m = idx >> 6, k = idx & 63;
  int s = m & (NS - 1);
  float v = 0.0f;
  if (k < 6)       v = x[m * 6 + k];
  else if (k < 38) v = xc[s * NG + (k - 6)];
  xp[idx] = (_Float16)v;
}

// W1^T padded: w1t[n*K1PAD + k] = W1[k, n] (k<38), else 0
__global__ void k_prep_w1t(const float* __restrict__ w1, _Float16* __restrict__ w1t) {
  int idx = blockIdx.x * 256 + threadIdx.x;          // HID*K1PAD
  int n = idx >> 6, k = idx & 63;
  w1t[idx] = (_Float16)((k < 38) ? w1[k * HID + n] : 0.0f);
}

// W2^T: w2t[n*HID + k] = W2[k, n]
__global__ void k_prep_w2t(const float* __restrict__ w2, _Float16* __restrict__ w2t) {
  int idx = blockIdx.x * 256 + threadIdx.x;          // N2*HID
  int n = idx >> 8, k = idx & 255;
  w2t[idx] = (_Float16)w2[k * N2 + n];
}

// ---------------- fcT GEMM 1: H1 = tanh(xcT @ W1 + b1), WMMA ----------------
// wave-task = 16x64 output strip; 5840 M-tiles x 4 N-strips = 23360 waves
__global__ void k_fcT1(const _Float16* __restrict__ xp, const _Float16* __restrict__ w1t,
                       const float* __restrict__ b1, _Float16* __restrict__ h1) {
  int wave = (blockIdx.x * 256 + threadIdx.x) >> 5;
  int lane = threadIdx.x & 31;
  int m0 = (wave >> 2) * 16;
  int n0 = (wave & 3) * 64;
  int mrow = m0 + (lane & 15);
  int kbA = (lane >> 4) * 8;
  int kbB = (lane >> 4) * 8;

  v8f acc[4];
#pragma unroll
  for (int t = 0; t < 4; ++t)
#pragma unroll
    for (int e = 0; e < 8; ++e) acc[t][e] = 0.0f;

#pragma unroll
  for (int kc = 0; kc < 2; ++kc) {
    v16h a = ld_a_frag(xp + mrow * K1PAD + kc * 32 + kbA);
#pragma unroll
    for (int t = 0; t < 4; ++t) {
      int n = n0 + t * 16 + (lane & 15);
      v16h b = ld_b_frag(w1t + n * K1PAD + kc * 32 + kbB);
      acc[t] = __builtin_amdgcn_wmma_f32_16x16x32_f16(
          false, a, false, b, (short)0, acc[t], false, false);
    }
  }

  int mbase = m0 + (lane >> 4) * 8;
#pragma unroll
  for (int t = 0; t < 4; ++t) {
    int n = n0 + t * 16 + (lane & 15);
    float bias = b1[n];
#pragma unroll
    for (int r = 0; r < 8; ++r)
      h1[(mbase + r) * HID + n] = (_Float16)tanhf(acc[t][r] + bias);
  }
}

// ---------------- fcT GEMM 2: V = H1 @ W2 + b2; fused activations ----------------
// columns [0,256): vi -> fl = P*(1-vf)*vi ; [256,512): ve -> fev = E*relu(v)*2 ;
// [512,768): vm -> fm = exp(v).  5840 M-tiles x 12 N-strips = 70080 waves.
__global__ void k_fcT2(const _Float16* __restrict__ h1, const _Float16* __restrict__ w2t,
                       const float* __restrict__ b2, const float* __restrict__ x,
                       _Float16* __restrict__ fl, _Float16* __restrict__ fev,
                       _Float16* __restrict__ fm) {
  int wave = (blockIdx.x * 256 + threadIdx.x) >> 5;
  int lane = threadIdx.x & 31;
  int m0 = (wave / 12) * 16;
  int n0 = (wave % 12) * 64;
  int mrow = m0 + (lane & 15);
  int kbA = (lane >> 4) * 8;
  int kbB = (lane >> 4) * 8;

  v8f acc[4];
#pragma unroll
  for (int t = 0; t < 4; ++t)
#pragma unroll
    for (int e = 0; e < 8; ++e) acc[t][e] = 0.0f;

  for (int kc = 0; kc < 8; ++kc) {
    v16h a = ld_a_frag(h1 + mrow * HID + kc * 32 + kbA);
#pragma unroll
    for (int t = 0; t < 4; ++t) {
      int n = n0 + t * 16 + (lane & 15);
      v16h b = ld_b_frag(w2t + n * HID + kc * 32 + kbB);
      acc[t] = __builtin_amdgcn_wmma_f32_16x16x32_f16(
          false, a, false, b, (short)0, acc[t], false, false);
    }
  }

  int mbase = m0 + (lane >> 4) * 8;
#pragma unroll
  for (int t = 0; t < 4; ++t) {
    int c = n0 + t * 16 + (lane & 15);
    float bias = b2[c];
    int h = c & (NH - 1);
#pragma unroll
    for (int r = 0; r < 8; ++r) {
      int m = mbase + r;
      float val = acc[t][r] + bias;
      const float* xr = x + m * 6;
      if (c < NH) {
        float vf = snow_frac(xr[2], xr[3]);
        float vi = fminf(fmaxf(val * (1.0f / 3.0f) + 0.5f, 0.0f), 1.0f);
        fl[m * NH + h] = (_Float16)(xr[0] * (1.0f - vf) * vi);
      } else if (c < 2 * NH) {
        float ve = fmaxf(val, 0.0f) * 2.0f;
        fev[m * NH + h] = (_Float16)(xr[1] * ve);
      } else {
        fm[m * NH + h] = (_Float16)expf(val);
      }
    }
  }
}

// ---------------- small MLPs (fcW / fcR), VALU ----------------
__global__ void k_mlp_h(const float* __restrict__ xc, const float* __restrict__ w1,
                        const float* __restrict__ b1, float* __restrict__ H) {
  int idx = blockIdx.x * 256 + threadIdx.x;          // NS*HID
  int s = idx >> 8, j = idx & 255;
  float acc = b1[j];
#pragma unroll
  for (int g = 0; g < NG; ++g) acc += xc[s * NG + g] * w1[g * HID + j];
  H[idx] = tanhf(acc);
}

__global__ void k_mlp_out(const float* __restrict__ H, const float* __restrict__ w2,
                          const float* __restrict__ b2, float* __restrict__ out,
                          int N, int do_relu) {
  int idx = blockIdx.x * 256 + threadIdx.x;          // NS*N
  int s = idx / N, o = idx % N;
  float acc = b2[o];
  for (int j = 0; j < HID; ++j) acc += H[s * HID + j] * w2[j * N + o];
  if (do_relu) acc = fmaxf(acc, 0.0f);
  out[idx] = acc;
}

// ---------------- per-basin params + softmax(ga) ----------------
__global__ void k_paramsW(const float* __restrict__ w, float* kp, float* ks, float* kg,
                          float* gp, float* gL, float* qb, float* ga) {
  __shared__ float red[256];
  int s = blockIdx.x, h = threadIdx.x;
  const float* wr = w + s * (NH * 7);
  int i = s * NH + h;
  kp[i] = sigm(wr[h]);
  ks[i] = sigm(wr[NH + h]);
  kg[i] = sigm(wr[2 * NH + h]) * 0.1f;
  gp[i] = sigm(wr[3 * NH + h]);
  float e = expf(wr[4 * NH + h]);
  gL[i] = e * e;
  qb[i] = fmaxf(wr[5 * NH + h], 0.0f);
  float z = wr[6 * NH + h];
  red[h] = z; __syncthreads();
  for (int o = 128; o > 0; o >>= 1) { if (h < o) red[h] = fmaxf(red[h], red[h + o]); __syncthreads(); }
  float mx = red[0]; __syncthreads();
  float ez = expf(z - mx);
  red[h] = ez; __syncthreads();
  for (int o = 128; o > 0; o >>= 1) { if (h < o) red[h] += red[h + o]; __syncthreads(); }
  ga[i] = ez / red[0];
}

// routing softmax over NR taps
__global__ void k_paramsR(const float* __restrict__ rf, float* __restrict__ r) {
  int idx = blockIdx.x * 256 + threadIdx.x;          // NS*NH
  const float* p = rf + idx * NR;
  float v[NR], mx = -1e30f;
#pragma unroll
  for (int j = 0; j < NR; ++j) { v[j] = p[j]; mx = fmaxf(mx, v[j]); }
  float sum = 0.0f;
#pragma unroll
  for (int j = 0; j < NR; ++j) { v[j] = expf(v[j] - mx); sum += v[j]; }
  float inv = 1.0f / sum;
#pragma unroll
  for (int j = 0; j < NR; ++j) r[idx * NR + j] = v[j] * inv;
}

// ---------------- fused sequential scan + 16-tap conv + nh-reduction ----------------
// 1 block per station (256 threads = 256 hidden units = 8 wave32).
__global__ void k_scan(const float* __restrict__ x, const _Float16* __restrict__ fl,
                       const _Float16* __restrict__ fev, const _Float16* __restrict__ fm,
                       const float* __restrict__ kp_, const float* __restrict__ ks_,
                       const float* __restrict__ kg_, const float* __restrict__ gp_,
                       const float* __restrict__ gL_, const float* __restrict__ qb_,
                       const float* __restrict__ ga_, const float* __restrict__ r_,
                       float* __restrict__ out) {
  __shared__ float red[8];
  int s = blockIdx.x, h = threadIdx.x;
  int i = s * NH + h;
  float kp = kp_[i], ks = ks_[i], kg = kg_[i], gp = gp_[i];
  float gL = gL_[i], qb = qb_[i], ga = ga_[i];
  float rg[NR];
#pragma unroll
  for (int j = 0; j < NR; ++j) rg[j] = ga * r_[i * NR + j];  // fold ga into taps
  float Sf = 0.0f, Ss = 0.0f, Sg = 0.0f;
  float A[NR];
#pragma unroll
  for (int j = 0; j < NR; ++j) A[j] = 0.0f;
  int lane = h & 31, wid = h >> 5;

  for (int t = 0; t < NT; ++t) {
    int m = t * NS + s;
    const float* xr = x + m * 6;
    float fs = xr[0] * snow_frac(xr[2], xr[3]);     // Ps = P*vf (uniform per block)
    int off = m * NH + h;
    float flv = (float)fl[off], fevv = (float)fev[off], fmv = (float)fm[off];

    float qf = fminf(Sf + fs, fmv);
    Sf = fmaxf(Sf + fs - fmv, 0.0f);
    float H = fmaxf(Ss + flv + qf - fevv, 0.0f);
    float qp = fmaxf(kp * (H - gL), 0.0f);
    float qsa = ks * fminf(H, gL);
    Ss = H - qp - qsa;
    float qsg = qsa * gp;
    float qs = qsa * (1.0f - gp);
    float qg = kg * (Sg + qsg) + qb;
    Sg = (1.0f - kg) * (Sg + qsg) - qb;
    float Q = qp + qs + qg;

#pragma unroll
    for (int j = 0; j < NR; ++j) A[j] += rg[j] * Q;

    if (t >= NR - 1) {                               // finalize y[t-15], reduce over nh
      float y = A[NR - 1];
#pragma unroll
      for (int o = 16; o > 0; o >>= 1) y += __shfl_xor(y, o, 32);
      if (lane == 0) red[wid] = y;
      __syncthreads();
      if (h == 0) {
        float tot = 0.0f;
#pragma unroll
        for (int w = 0; w < 8; ++w) tot += red[w];
        out[(t - (NR - 1)) * NS + s] = tot;
      }
      __syncthreads();
    }
#pragma unroll
    for (int j = NR - 1; j > 0; --j) A[j] = A[j - 1]; // shift register
    A[0] = 0.0f;
  }
}

// ---------------- launcher ----------------
extern "C" void kernel_launch(void* const* d_in, const int* in_sizes, int n_in,
                              void* d_out, int out_size, void* d_ws, size_t ws_size,
                              hipStream_t stream) {
  const float* x      = (const float*)d_in[0];
  const float* xc     = (const float*)d_in[1];
  const float* fcW_w1 = (const float*)d_in[2];
  const float* fcW_b1 = (const float*)d_in[3];
  const float* fcW_w2 = (const float*)d_in[4];
  const float* fcW_b2 = (const float*)d_in[5];
  const float* fcT_w1 = (const float*)d_in[6];
  const float* fcT_b1 = (const float*)d_in[7];
  const float* fcT_w2 = (const float*)d_in[8];
  const float* fcT_b2 = (const float*)d_in[9];
  const float* fcR_w1 = (const float*)d_in[10];
  const float* fcR_b1 = (const float*)d_in[11];
  const float* fcR_w2 = (const float*)d_in[12];
  const float* fcR_b2 = (const float*)d_in[13];
  float* out = (float*)d_out;

  char* ws = (char*)d_ws;
  size_t off = 0;
  auto take = [&](size_t bytes) -> void* {
    void* p = ws + off;
    off += (bytes + 255) & ~(size_t)255;
    return p;
  };
  _Float16* xp  = (_Float16*)take((size_t)M_ROWS * K1PAD * 2);
  _Float16* w1t = (_Float16*)take((size_t)HID * K1PAD * 2);
  _Float16* w2t = (_Float16*)take((size_t)N2 * HID * 2);
  _Float16* h1  = (_Float16*)take((size_t)M_ROWS * HID * 2);
  _Float16* fl  = (_Float16*)take((size_t)M_ROWS * NH * 2);
  _Float16* fev = (_Float16*)take((size_t)M_ROWS * NH * 2);
  _Float16* fm  = (_Float16*)take((size_t)M_ROWS * NH * 2);
  float* Hbuf = (float*)take((size_t)NS * HID * 4);
  float* wraw = (float*)take((size_t)NS * NH * 7 * 4);
  float* rf   = (float*)take((size_t)NS * NH * NR * 4);
  float* kp   = (float*)take((size_t)NS * NH * 4);
  float* ks   = (float*)take((size_t)NS * NH * 4);
  float* kg   = (float*)take((size_t)NS * NH * 4);
  float* gp   = (float*)take((size_t)NS * NH * 4);
  float* gL   = (float*)take((size_t)NS * NH * 4);
  float* qb   = (float*)take((size_t)NS * NH * 4);
  float* ga   = (float*)take((size_t)NS * NH * 4);
  float* r    = (float*)take((size_t)NS * NH * NR * 4);

  // prep: pack/transpose into f16
  k_prep_xcT<<<M_ROWS * K1PAD / 256, 256, 0, stream>>>(x, xc, xp);
  k_prep_w1t<<<HID * K1PAD / 256, 256, 0, stream>>>(fcT_w1, w1t);
  k_prep_w2t<<<N2 * HID / 256, 256, 0, stream>>>(fcT_w2, w2t);

  // fcT: the two big WMMA GEMMs
  k_fcT1<<<(M_ROWS / 16) * 4 / 8, 256, 0, stream>>>(xp, w1t, fcT_b1, h1);
  k_fcT2<<<(M_ROWS / 16) * 12 / 8, 256, 0, stream>>>(h1, w2t, fcT_b2, x, fl, fev, fm);

  // fcW
  k_mlp_h<<<NS * HID / 256, 256, 0, stream>>>(xc, fcW_w1, fcW_b1, Hbuf);
  k_mlp_out<<<NS * NH * 7 / 256, 256, 0, stream>>>(Hbuf, fcW_w2, fcW_b2, wraw, NH * 7, 0);
  k_paramsW<<<NS, 256, 0, stream>>>(wraw, kp, ks, kg, gp, gL, qb, ga);

  // fcR
  k_mlp_h<<<NS * HID / 256, 256, 0, stream>>>(xc, fcR_w1, fcR_b1, Hbuf);
  k_mlp_out<<<NS * NH * NR / 256, 256, 0, stream>>>(Hbuf, fcR_w2, fcR_b2, rf, NH * NR, 1);
  k_paramsR<<<NS * NH / 256, 256, 0, stream>>>(rf, r);

  // fused scan + conv + reduction -> out [350, 256]
  k_scan<<<NS, 256, 0, stream>>>(x, fl, fev, fm, kp, ks, kg, gp, gL, qb, ga, r, out);
}
